// MHA_83296595738764
// MI455X (gfx1250) — matrix operand order repaired and checked by reading, not compiled, not run on hardware
//
#include <hip/hip_runtime.h>
#include <hip/hip_bf16.h>
#include <cstdint>

// ---------------------------------------------------------------------------
// Transformer-XL relative-position attention, MI455X (gfx1250, wave32, WMMA)
// All matmuls via v_wmma_f32_16x16x32_bf16; f32 accumulation everywhere.
// GEMM tiles are staged by the Tensor Data Mover (tensor_load_to_lds) with
// double-buffered LDS so the DMA overlaps the WMMA stream; fragment loads are
// b128-vectorized per the ISA 7.12.2 lane layouts.
// ---------------------------------------------------------------------------

typedef __bf16 bf16_t;
typedef __attribute__((ext_vector_type(16))) __bf16 v16bf;
typedef __attribute__((ext_vector_type(8)))  float  v8f;
typedef unsigned int v4u __attribute__((ext_vector_type(4)));
typedef int          v4i __attribute__((ext_vector_type(4)));
typedef int          v8i __attribute__((ext_vector_type(8)));

#define SEQ   2048
#define HEADS 16
#define DK    64
#define CDIM  1024
#define BATCH 2
#define MTOK  (BATCH * SEQ)      // 4096 token rows

#if defined(__has_builtin)
#  if __has_builtin(__builtin_amdgcn_tensor_load_to_lds)
#    define USE_TDM 1
#  endif
#endif
#ifndef USE_TDM
#  define USE_TDM 0
#endif

// ---------------- bf16 helpers (round-to-nearest-even) ----------------------
__device__ __forceinline__ unsigned f2bf_bits(float f) {
    unsigned u = __builtin_bit_cast(unsigned, f);
    return (u + 0x7FFFu + ((u >> 16) & 1u)) >> 16;
}
__device__ __forceinline__ bf16_t f2bf(float f) {
    unsigned short h = (unsigned short)f2bf_bits(f);
    return __builtin_bit_cast(bf16_t, h);
}
__device__ __forceinline__ float bf2f(bf16_t b) {
    unsigned u = ((unsigned)__builtin_bit_cast(unsigned short, b)) << 16;
    return __builtin_bit_cast(float, u);
}
__device__ __forceinline__ v8f vzero8() {
    v8f z;
#pragma unroll
    for (int i = 0; i < 8; ++i) z[i] = 0.0f;
    return z;
}
__device__ __forceinline__ void wait_tensor0() {
#if defined(__has_builtin)
#  if __has_builtin(__builtin_amdgcn_s_wait_tensorcnt)
    __builtin_amdgcn_s_wait_tensorcnt((short)0);
#  endif
#endif
}

// ---------------- WMMA fragment loaders (ISA 7.12.2 layouts) -----------------
// A 16x32 bf16: lane half kh = 8*(lane>=16); per lane two contiguous 16B runs
// at element offsets kh and 16+kh of row (lane&15).
__device__ __forceinline__ v16bf load_frag_a(const bf16_t* base, int stride, int lane) {
    int r  = lane & 15;
    int kh = (lane >> 4) << 3;
    union { v16bf v; uint4 q[2]; } f;
    const bf16_t* p = base + (size_t)r * stride + kh;
    f.q[0] = *(const uint4*)(p);
    f.q[1] = *(const uint4*)(p + 16);
    return f.v;
}
// B 32x16 bf16 stored as N x K: lane half kh = 8*(lane>=16); per lane two
// contiguous 16B runs at element offsets kh and 16+kh of row (lane&15).
__device__ __forceinline__ v16bf load_frag_b(const bf16_t* base, int stride, int lane) {
    int n  = lane & 15;
    int kh = (lane >> 4) << 3;
    union { v16bf v; uint4 q[2]; } f;
    const bf16_t* p = base + (size_t)n * stride + kh;
    f.q[0] = *(const uint4*)(p);
    f.q[1] = *(const uint4*)(p + 16);
    return f.v;
}

// ---------------- Tensor Data Mover 2D tile load ----------------------------
__device__ __forceinline__ unsigned lds_addr_of(const void* p) {
    // generic pointer to LDS: low 32 bits are the LDS byte address (ISA 10.2)
    return (unsigned)(uintptr_t)p;
}

#if USE_TDM
// Load a tile_d1 x tile_d0 (rows x elements) bf16 tile whose top-left corner
// is at gaddr inside a tensor with row stride stride_elems. LDS destination is
// padded: pad_amount+1 DWORDs inserted every 2^(pad_interval+1) DWORDs.
__device__ __forceinline__ void tdm_load_tile(unsigned lds_addr, const void* gaddr,
                                              unsigned tensor_d0, unsigned tensor_d1,
                                              unsigned tile_d0, unsigned tile_d1,
                                              unsigned stride_elems,
                                              unsigned pad_interval, unsigned pad_amount) {
    unsigned long long ga = (unsigned long long)(uintptr_t)gaddr;
    v4u g0;
    g0[0] = 1u;                                          // count=1, user descriptor
    g0[1] = lds_addr;                                    // lds_addr (bytes)
    g0[2] = (unsigned)(ga & 0xFFFFFFFFu);                // global_addr[31:0]
    g0[3] = (unsigned)((ga >> 32) & 0x1FFFFFFu) | (2u << 30);   // addr[56:32] | type=2
    v8i g1;
    g1[0] = (int)((1u << 16) |                           // data_size = 2 bytes
                  (1u << 20) |                           // pad_enable
                  (pad_interval << 22) | (pad_amount << 25));
    g1[1] = (int)((tensor_d0 & 0xFFFFu) << 16);          // abar=0 | tensor_dim0 lo
    g1[2] = (int)(((tensor_d0 >> 16) & 0xFFFFu) | ((tensor_d1 & 0xFFFFu) << 16));
    g1[3] = (int)(((tensor_d1 >> 16) & 0xFFFFu) | ((tile_d0 & 0xFFFFu) << 16));
    g1[4] = (int)(tile_d1 & 0xFFFFu);                    // tile_dim1 | tile_dim2=0
    g1[5] = (int)stride_elems;                           // tensor_dim0_stride lo32
    g1[6] = 0;                                           // stride hi | dim1_stride lo
    g1[7] = 0;
    v4i z4; z4[0] = z4[1] = z4[2] = z4[3] = 0;
#if __clang_major__ >= 23
    v8i z8;
#pragma unroll
    for (int i = 0; i < 8; ++i) z8[i] = 0;
    __builtin_amdgcn_tensor_load_to_lds(g0, g1, z4, z4, z8, 0);
#else
    __builtin_amdgcn_tensor_load_to_lds(g0, g1, z4, z4, 0);
#endif
}
#endif

// ---------------- small prep kernels ----------------------------------------
__global__ void cast_bf16_kernel(const float* __restrict__ src,
                                 bf16_t* __restrict__ dst, int n) {
    int i = blockIdx.x * blockDim.x + threadIdx.x;
    if (i < n) dst[i] = f2bf(src[i]);
}

// dst[n*rows + k] = src[k*cols + n]   (store W^T in bf16 so B-tiles are NxK)
__global__ void transpose_cast_kernel(const float* __restrict__ src,
                                      bf16_t* __restrict__ dst,
                                      int rows, int cols) {
    int i = blockIdx.x * blockDim.x + threadIdx.x;
    if (i >= rows * cols) return;
    int k = i / cols, n = i - k * cols;
    dst[(size_t)n * rows + k] = f2bf(src[i]);
}

// r[j,d] sinusoid (pos = S-1-j) in bf16, and term4[h,j] = v[h]·r[j] in f32
__global__ void relpos_kernel(const float* __restrict__ v,
                              bf16_t* __restrict__ rbf,
                              float* __restrict__ term4) {
    int j = blockIdx.x * blockDim.x + threadIdx.x;
    if (j >= SEQ) return;
    float pos = (float)(SEQ - 1 - j);
    float r[DK];
    float lg = logf(10000.0f);
#pragma unroll 4
    for (int i = 0; i < 32; ++i) {
        float freq = expf(-((float)(2 * i) / 64.0f) * lg);
        float ang  = pos * freq;
        r[i]      = sinf(ang);
        r[32 + i] = cosf(ang);
    }
#pragma unroll 4
    for (int d = 0; d < DK; ++d) rbf[(size_t)j * DK + d] = f2bf(r[d]);
    for (int h = 0; h < HEADS; ++h) {
        float s = 0.0f;
#pragma unroll 4
        for (int d = 0; d < DK; ++d) s += v[h * DK + d] * r[d];
        term4[(size_t)h * SEQ + j] = s;
    }
}

// term3[b,h,k] = u[h]·K[b,h,k,:]
__global__ void term3_kernel(const float* __restrict__ u,
                             const bf16_t* __restrict__ Kb,
                             float* __restrict__ term3) {
    int idx = blockIdx.x * blockDim.x + threadIdx.x;   // (b*H+h)*S + k
    if (idx >= BATCH * HEADS * SEQ) return;
    int h = (idx >> 11) & (HEADS - 1);
    const bf16_t* kr = Kb + (size_t)idx * DK;
    const float*  ur = u + h * DK;
    float s = 0.0f;
#pragma unroll 4
    for (int d = 0; d < DK; ++d) s += ur[d] * bf2f(kr[d]);
    term3[idx] = s;
}

// ---------------- tiled bf16 GEMM: C[M,N] = A[M,K] * BT[N,K]^T ---------------
// 128x128 block tile, BK=32, 256 threads = 8 waves (2x4), wave tile 64x32.
// TDM double-buffers the LDS tiles so DMA overlaps the WMMA stream.
#define BM 128
#define BN 128
#define BK 32
#define LDSK 40   // padded row stride: 80B rows (16B aligned, bank-spread)

#define EPI_F32_BIAS 0   // out f32 row-major [M,N] + bias[n]
#define EPI_QK       1   // out bf16 [b,h,s,d]
#define EPI_VT       2   // out bf16 [b,h,d,s]

__global__ void gemm_bf16_kernel(const bf16_t* __restrict__ A,
                                 const bf16_t* __restrict__ BT,
                                 bf16_t* __restrict__ outb,
                                 float* __restrict__ outf,
                                 const float* __restrict__ bias,
                                 int M, int N, int K, int epi) {
    __shared__ bf16_t sA[2][BM * LDSK];
    __shared__ bf16_t sB[2][BN * LDSK];

    int tid  = threadIdx.x;
    int lane = tid & 31;
    int w    = tid >> 5;           // 0..7
    int wm   = w >> 2;             // 0..1  (rows)
    int wn   = w & 3;              // 0..3  (cols)

    int ntn = N >> 7;
    int m0  = (blockIdx.x / ntn) * BM;
    int n0  = (blockIdx.x % ntn) * BN;

    v8f acc[4][2];
#pragma unroll
    for (int i = 0; i < 4; ++i)
#pragma unroll
        for (int j = 0; j < 2; ++j) acc[i][j] = vzero8();

#if USE_TDM
    // prologue: DMA tile 0 into buffer 0
    if (w == 0) {
        tdm_load_tile(lds_addr_of(sA[0]), A + (size_t)m0 * K,
                      (unsigned)K, (unsigned)M, BK, BM, (unsigned)K, 3u, 3u);
        tdm_load_tile(lds_addr_of(sB[0]), BT + (size_t)n0 * K,
                      (unsigned)K, (unsigned)N, BK, BN, (unsigned)K, 3u, 3u);
        wait_tensor0();
    }
    __syncthreads();
#endif

    int p = 0;
    for (int k0 = 0; k0 < K; k0 += BK) {
#if USE_TDM
        // kick off the DMA for the next tile into the spare buffer; it runs
        // concurrently with this iteration's 64 WMMAs (TENSORcnt-tracked).
        if (w == 0 && k0 + BK < K) {
            tdm_load_tile(lds_addr_of(sA[p ^ 1]), A + (size_t)m0 * K + k0 + BK,
                          (unsigned)K, (unsigned)M, BK, BM, (unsigned)K, 3u, 3u);
            tdm_load_tile(lds_addr_of(sB[p ^ 1]), BT + (size_t)n0 * K + k0 + BK,
                          (unsigned)K, (unsigned)N, BK, BN, (unsigned)K, 3u, 3u);
        }
#else
        __syncthreads();
        // manual staging fallback: 128x32 bf16 tiles of A and BT
#pragma unroll
        for (int i = tid; i < (BM * BK / 4); i += 256) {
            int row = i >> 3;
            int c4  = (i & 7) << 2;
            *(uint2*)&sA[0][row * LDSK + c4] =
                *(const uint2*)&A[(size_t)(m0 + row) * K + k0 + c4];
            *(uint2*)&sB[0][row * LDSK + c4] =
                *(const uint2*)&BT[(size_t)(n0 + row) * K + k0 + c4];
            if (k0 + BK < K) {
                __builtin_prefetch(&A[(size_t)(m0 + row) * K + k0 + BK + c4], 0, 1);
                __builtin_prefetch(&BT[(size_t)(n0 + row) * K + k0 + BK + c4], 0, 1);
            }
        }
        __syncthreads();
#endif
        const bf16_t* curA = sA[p];
        const bf16_t* curB = sB[p];

        v16bf af[4];
#pragma unroll
        for (int fm = 0; fm < 4; ++fm)
            af[fm] = load_frag_a(&curA[(wm * 64 + fm * 16) * LDSK], LDSK, lane);
#pragma unroll
        for (int fn = 0; fn < 2; ++fn) {
            v16bf bf = load_frag_b(&curB[(wn * 32 + fn * 16) * LDSK], LDSK, lane);
#pragma unroll
            for (int fm = 0; fm < 4; ++fm) {
                acc[fm][fn] = __builtin_amdgcn_wmma_f32_16x16x32_bf16(
                    false, af[fm], false, bf, (short)0, acc[fm][fn], false, false);
            }
        }

#if USE_TDM
        __syncthreads();                 // all waves done reading buffer p
        if (w == 0) wait_tensor0();      // next tile fully landed in p^1
        __syncthreads();                 // publish buffer p^1
        p ^= 1;
#endif
    }

    // epilogue: C/D layout -> (m = j + 8*(lane>=16), n = lane&15)
    int mb = (lane >> 4) << 3;
    int nl = lane & 15;
#pragma unroll
    for (int fm = 0; fm < 4; ++fm) {
#pragma unroll
        for (int fn = 0; fn < 2; ++fn) {
#pragma unroll
            for (int j = 0; j < 8; ++j) {
                int m = m0 + wm * 64 + fm * 16 + j + mb;
                int n = n0 + wn * 32 + fn * 16 + nl;
                float val = acc[fm][fn][j];
                if (epi == EPI_F32_BIAS) {
                    outf[(size_t)m * N + n] = val + bias[n];
                } else {
                    int bq = m >> 11, s = m & (SEQ - 1);
                    int h  = n >> 6,  d = n & (DK - 1);
                    if (epi == EPI_QK)
                        outb[((size_t)(bq * HEADS + h) * SEQ + s) * DK + d] = f2bf(val);
                    else
                        outb[((size_t)(bq * HEADS + h) * DK + d) * SEQ + s] = f2bf(val);
                }
            }
        }
    }
}

// ---------------- fused TXL attention (per b, h, 16-row q-tile) --------------
// LDS: bd[17][S] f32  | scores[16][S] f32 ; probs(bf16) overlays bd region,
// PV partials overlay the scores region once softmax is done.
#define BD_BYTES   (17 * SEQ * 4)
#define SC_BYTES   (16 * SEQ * 4)
#define ATTN_SMEM  (BD_BYTES + SC_BYTES)

__global__ void attn_kernel(const bf16_t* __restrict__ Qb,
                            const bf16_t* __restrict__ Kb,
                            const bf16_t* __restrict__ Vtb,
                            const bf16_t* __restrict__ rbf,
                            const float*  __restrict__ term3,
                            const float*  __restrict__ term4,
                            bf16_t* __restrict__ O) {
    extern __shared__ char smem[];
    float*  bd      = (float*)smem;                 // [17][SEQ]
    float*  scores  = (float*)(smem + BD_BYTES);    // [16][SEQ]
    bf16_t* probs   = (bf16_t*)smem;                // overlays bd  [16][SEQ]
    float*  partial = (float*)(smem + BD_BYTES);    // overlays scores [8][256]

    int tid  = threadIdx.x;
    int lane = tid & 31;
    int w    = tid >> 5;

    int qt = blockIdx.x & 127;
    int h  = (blockIdx.x >> 7) & (HEADS - 1);
    int b  = blockIdx.x >> 11;
    int q0 = qt * 16;

    const bf16_t* Qrow = Qb + ((size_t)(b * HEADS + h) * SEQ + q0) * DK;
    const bf16_t* Kbh  = Kb + (size_t)(b * HEADS + h) * SEQ * DK;
    const float*  t3   = term3 + (size_t)(b * HEADS + h) * SEQ;
    const float*  t4   = term4 + (size_t)h * SEQ;

    // Q tile fragments (shared by term1 and term2)
    v16bf aq0 = load_frag_a(Qrow,      DK, lane);
    v16bf aq1 = load_frag_a(Qrow + 32, DK, lane);

    int mb = (lane >> 4) << 3;
    int nl = lane & 15;

    // ---- phase 1: BD = Q·r^T + term4  (rows 0..15 via WMMA) ----------------
    for (int nt = w; nt < SEQ / 16; nt += 8) {
        int n0 = nt * 16;
        v16bf b0 = load_frag_b(rbf + (size_t)n0 * DK,      DK, lane);
        v16bf b1 = load_frag_b(rbf + (size_t)n0 * DK + 32, DK, lane);
        v8f a = vzero8();
        a = __builtin_amdgcn_wmma_f32_16x16x32_bf16(false, aq0, false, b0, (short)0, a, false, false);
        a = __builtin_amdgcn_wmma_f32_16x16x32_bf16(false, aq1, false, b1, (short)0, a, false, false);
#pragma unroll
        for (int j = 0; j < 8; ++j) {
            int m = j + mb, col = n0 + nl;
            bd[m * SEQ + col] = a[j] + t4[col];
        }
    }
    // row 16 (needed for the wrap part of the rel-shift), VALU
    {
        const bf16_t* qx = Qrow + 16 * DK;   // row q0+16 (exists unless last tile)
        bool valid = (q0 + 16) < SEQ;
        for (int c = tid; c < SEQ; c += 256) {
            float s = 0.0f;
            if (valid) {
                const bf16_t* rr = rbf + (size_t)c * DK;
#pragma unroll 4
                for (int d = 0; d < DK; ++d) s += bf2f(qx[d]) * bf2f(rr[d]);
                s += t4[c];
            }
            bd[16 * SEQ + c] = s;
        }
    }
    __syncthreads();

    // ---- phase 2: scores = (Q·K^T + term3 + shift(BD)) / sqrt(dk) ----------
    for (int nt = w; nt < SEQ / 16; nt += 8) {
        int n0 = nt * 16;
        v16bf b0 = load_frag_b(Kbh + (size_t)n0 * DK,      DK, lane);
        v16bf b1 = load_frag_b(Kbh + (size_t)n0 * DK + 32, DK, lane);
        v8f a = vzero8();
        a = __builtin_amdgcn_wmma_f32_16x16x32_bf16(false, aq0, false, b0, (short)0, a, false, false);
        a = __builtin_amdgcn_wmma_f32_16x16x32_bf16(false, aq1, false, b1, (short)0, a, false, false);
#pragma unroll
        for (int j = 0; j < 8; ++j) {
            int m = j + mb;
            int k = n0 + nl;
            int qp = q0 + m;
            unsigned f = (unsigned)(qp + 1) * SEQ + (unsigned)k;  // pad-reshape-slice shift
            unsigned q = f / (SEQ + 1);
            unsigned t = f - q * (SEQ + 1);
            float sh = (t == 0u) ? 0.0f : bd[(q - (unsigned)q0) * SEQ + (t - 1)];
            scores[m * SEQ + k] = (a[j] + t3[k] + sh) * 0.125f;  // 1/sqrt(64)
        }
    }
    __syncthreads();

    // ---- phase 3: row softmax -> probs (bf16, overlays bd) -----------------
    for (int rr = 2 * w; rr < 2 * w + 2; ++rr) {
        float* srow = scores + rr * SEQ;
        float mx = -3.4e38f;
        for (int c = lane; c < SEQ; c += 32) mx = fmaxf(mx, srow[c]);
#pragma unroll
        for (int off = 16; off >= 1; off >>= 1) mx = fmaxf(mx, __shfl_xor(mx, off, 32));
        float sum = 0.0f;
        for (int c = lane; c < SEQ; c += 32) {
            float e = expf(srow[c] - mx);
            srow[c] = e;
            sum += e;
        }
#pragma unroll
        for (int off = 16; off >= 1; off >>= 1) sum += __shfl_xor(sum, off, 32);
        float inv = 1.0f / sum;
        for (int c = lane; c < SEQ; c += 32)
            probs[rr * SEQ + c] = f2bf(srow[c] * inv);
    }
    __syncthreads();

    // ---- phase 4: O = P · V  (wave = (d-tile, k-half); partial reduce) -----
    {
        int nt = w >> 1, kh = w & 1, d0 = nt * 16;
        const bf16_t* Vrow = Vtb + ((size_t)(b * HEADS + h) * DK + d0) * SEQ;
        v8f a = vzero8();
        for (int ks = kh * (SEQ / 2); ks < kh * (SEQ / 2) + SEQ / 2; ks += 32) {
            v16bf ap = load_frag_a(probs + ks, SEQ, lane);
            v16bf bv = load_frag_b(Vrow + ks,  SEQ, lane);
            a = __builtin_amdgcn_wmma_f32_16x16x32_bf16(false, ap, false, bv, (short)0, a, false, false);
        }
#pragma unroll
        for (int j = 0; j < 8; ++j) partial[w * 256 + j * 32 + lane] = a[j];
    }
    __syncthreads();
    if (w < 4) {
        int d0 = w * 16;
#pragma unroll
        for (int j = 0; j < 8; ++j) {
            float s = partial[(2 * w) * 256 + j * 32 + lane] +
                      partial[(2 * w + 1) * 256 + j * 32 + lane];
            int m = j + mb, n = nl;
            O[((size_t)b * SEQ + q0 + m) * CDIM + h * DK + d0 + n] = f2bf(s);
        }
    }
}

// ---------------------------------------------------------------------------
extern "C" void kernel_launch(void* const* d_in, const int* in_sizes, int n_in,
                              void* d_out, int out_size, void* d_ws, size_t ws_size,
                              hipStream_t stream) {
    (void)in_sizes; (void)n_in; (void)out_size; (void)ws_size;
    const float* x  = (const float*)d_in[0];
    const float* Wq = (const float*)d_in[1];
    const float* Wk = (const float*)d_in[2];
    const float* Wv = (const float*)d_in[3];
    const float* u  = (const float*)d_in[4];
    const float* v  = (const float*)d_in[5];
    const float* Wo = (const float*)d_in[6];
    const float* bo = (const float*)d_in[7];
    float* out = (float*)d_out;

    char* ws = (char*)d_ws;
    size_t off = 0;
    auto alloc = [&](size_t bytes) { size_t o = off; off += (bytes + 255) & ~(size_t)255; return o; };
    bf16_t* xb   = (bf16_t*)(ws + alloc((size_t)MTOK * CDIM * 2));
    bf16_t* WqT  = (bf16_t*)(ws + alloc((size_t)CDIM * CDIM * 2));
    bf16_t* WkT  = (bf16_t*)(ws + alloc((size_t)CDIM * CDIM * 2));
    bf16_t* WvT  = (bf16_t*)(ws + alloc((size_t)CDIM * CDIM * 2));
    bf16_t* WoT  = (bf16_t*)(ws + alloc((size_t)CDIM * CDIM * 2));
    bf16_t* Qb   = (bf16_t*)(ws + alloc((size_t)MTOK * CDIM * 2));
    bf16_t* Kb   = (bf16_t*)(ws + alloc((size_t)MTOK * CDIM * 2));
    bf16_t* Vtb  = (bf16_t*)(ws + alloc((size_t)MTOK * CDIM * 2));
    bf16_t* Ob   = (bf16_t*)(ws + alloc((size_t)MTOK * CDIM * 2));
    bf16_t* rbf  = (bf16_t*)(ws + alloc((size_t)SEQ * DK * 2));
    float*  t4   = (float*) (ws + alloc((size_t)HEADS * SEQ * 4));
    float*  t3   = (float*) (ws + alloc((size_t)BATCH * HEADS * SEQ * 4));

    const int nx = MTOK * CDIM;
    cast_bf16_kernel<<<(nx + 255) / 256, 256, 0, stream>>>(x, xb, nx);
    const int nw = CDIM * CDIM;
    transpose_cast_kernel<<<(nw + 255) / 256, 256, 0, stream>>>(Wq, WqT, CDIM, CDIM);
    transpose_cast_kernel<<<(nw + 255) / 256, 256, 0, stream>>>(Wk, WkT, CDIM, CDIM);
    transpose_cast_kernel<<<(nw + 255) / 256, 256, 0, stream>>>(Wv, WvT, CDIM, CDIM);
    transpose_cast_kernel<<<(nw + 255) / 256, 256, 0, stream>>>(Wo, WoT, CDIM, CDIM);
    relpos_kernel<<<SEQ / 256, 256, 0, stream>>>(v, rbf, t4);

    dim3 gblk((MTOK / BM) * (CDIM / BN));
    gemm_bf16_kernel<<<gblk, 256, 0, stream>>>(xb, WqT, Qb,  nullptr, nullptr, MTOK, CDIM, CDIM, EPI_QK);
    gemm_bf16_kernel<<<gblk, 256, 0, stream>>>(xb, WkT, Kb,  nullptr, nullptr, MTOK, CDIM, CDIM, EPI_QK);
    gemm_bf16_kernel<<<gblk, 256, 0, stream>>>(xb, WvT, Vtb, nullptr, nullptr, MTOK, CDIM, CDIM, EPI_VT);

    const int n3 = BATCH * HEADS * SEQ;
    term3_kernel<<<(n3 + 255) / 256, 256, 0, stream>>>(u, Kb, t3);

    attn_kernel<<<BATCH * HEADS * (SEQ / 16), 256, ATTN_SMEM, stream>>>(
        Qb, Kb, Vtb, rbf, t3, t4, Ob);

    gemm_bf16_kernel<<<gblk, 256, 0, stream>>>(Ob, WoT, nullptr, out, bo, MTOK, CDIM, CDIM, EPI_F32_BIAS);
}
